// DecoderLayer_1941325218039
// MI455X (gfx1250) — compile-verified, matrix operand order checked by reading, not run on hardware
//
#include <hip/hip_runtime.h>
#include <hip/hip_bf16.h>

#define D_MODEL 512
#define N_HEADS 8
#define D_HEAD  64
#define D_INNER 2048
#define QLEN    2048
#define MLEN    2048
#define KLEN    (QLEN + MLEN)   // 4096
#define BATCH   2
#define NDH     (N_HEADS * D_HEAD)  // 512
#define MROWS   (QLEN * BATCH)      // 4096
#define CROWS   (KLEN * BATCH)      // 8192

#if __has_builtin(__builtin_amdgcn_global_load_async_to_lds_b128) && \
    __has_builtin(__builtin_amdgcn_s_wait_asynccnt)
#define HAVE_ASYNC_LDS 1
#else
#define HAVE_ASYNC_LDS 0
#endif

typedef _Float16 v16h __attribute__((ext_vector_type(16)));
typedef _Float16 v8h  __attribute__((ext_vector_type(8)));
typedef float    v8f  __attribute__((ext_vector_type(8)));
typedef int      v4i  __attribute__((ext_vector_type(4)));

#if HAVE_ASYNC_LDS
typedef __attribute__((address_space(1))) v4i* gptr4_t;  // global int4*
typedef __attribute__((address_space(3))) v4i* lptr4_t;  // LDS int4*
#endif

__device__ __forceinline__ v16h mk16(v8h lo, v8h hi) {
  union { v16h v; v8h h[2]; } u;
  u.h[0] = lo; u.h[1] = hi;
  return u.v;
}

// Load a 16x32 f16 WMMA fragment from a row-major matrix.
// Works for A (row = M) and for B of an NT GEMM (row = N), since the CDNA5
// 16-bit A layout puts row = lane&15 and K in two contiguous 8-element runs:
// lanes 0-15: K = {0..7, 16..23}; lanes 16-31: K = {8..15, 24..31}.
__device__ __forceinline__ v16h ld_frag(const _Float16* mat, int row0, int maxrow,
                                        int ld, int k0) {
  int lane = threadIdx.x & 31;
  int row = row0 + (lane & 15);
  row = row > maxrow ? maxrow : row;
  int kb = (lane & 16) ? 8 : 0;
  const _Float16* p = mat + (size_t)row * ld + k0 + kb;
  v8h lo = *(const v8h*)(p);
  v8h hi = *(const v8h*)(p + 16);
  return mk16(lo, hi);
}

__device__ __forceinline__ v8f wmma16(v16h a, v16h b, v8f c) {
  return __builtin_amdgcn_wmma_f32_16x16x32_f16(false, a, false, b, (short)0, c,
                                                false, false);
}

// ---------------------------------------------------------------- pack/cast
__global__ void k_cast(const float* __restrict__ s, _Float16* __restrict__ d, int n) {
  int i = blockIdx.x * blockDim.x + threadIdx.x;
  if (i < n) d[i] = (_Float16)s[i];
}

__global__ void k_transpose_cast(const float* __restrict__ s, _Float16* __restrict__ d,
                                 int R, int C) {  // s[R][C] -> d[C][R]
  int i = blockIdx.x * blockDim.x + threadIdx.x;
  if (i >= R * C) return;
  int r = i / C, c = i - r * C;
  d[(size_t)c * R + r] = (_Float16)s[i];
}

__global__ void k_pack_cat(const float* __restrict__ x, const float* __restrict__ mems,
                           _Float16* __restrict__ d) {
  int i = blockIdx.x * blockDim.x + threadIdx.x;  // over CROWS*D_MODEL
  if (i >= CROWS * D_MODEL) return;
  int c = i & (D_MODEL - 1);
  int row = i >> 9;                // j*BATCH + b
  int b = row & (BATCH - 1);
  int j = row >> 1;
  float val = (j < MLEN) ? mems[((size_t)j * BATCH + b) * D_MODEL + c]
                         : x[((size_t)(j - MLEN) * BATCH + b) * D_MODEL + c];
  d[i] = (_Float16)val;
}

__global__ void k_split_heads(const _Float16* __restrict__ qkv,
                              _Float16* __restrict__ q_h, _Float16* __restrict__ k_h,
                              _Float16* __restrict__ vT_h) {
  int i = blockIdx.x * blockDim.x + threadIdx.x;  // over KLEN*BATCH*NDH
  if (i >= KLEN * BATCH * NDH) return;
  int d = i & 63; int t = i >> 6;
  int h = t & 7;  t >>= 3;
  int b = t & 1;  int j = t >> 1;
  const _Float16* src = qkv + ((size_t)j * BATCH + b) * (3 * NDH);
  int bh = b * N_HEADS + h;
  k_h [((size_t)bh * KLEN + j) * D_HEAD + d] = src[NDH + h * D_HEAD + d];
  vT_h[((size_t)bh * D_HEAD + d) * KLEN + j] = src[2 * NDH + h * D_HEAD + d];
  if (j >= MLEN)
    q_h[((size_t)bh * QLEN + (j - MLEN)) * D_HEAD + d] = src[h * D_HEAD + d];
}

__global__ void k_split_rk(const _Float16* __restrict__ rkf, _Float16* __restrict__ rk_h) {
  int i = blockIdx.x * blockDim.x + threadIdx.x;  // over KLEN*NDH
  if (i >= KLEN * NDH) return;
  int d = i & 63; int t = i >> 6;
  int h = t & 7;  int j = t >> 3;
  rk_h[((size_t)h * KLEN + j) * D_HEAD + d] = rkf[(size_t)j * NDH + h * D_HEAD + d];
}

// ---------------------------------------------------------------- GEMM (NT)
// C[M,N] = A[M,K] @ B[N,K]^T (+bias)(+relu).
// One wave computes a 16x64 tile: one A fragment drives 4 WMMAs per K-step.
// The A tile is double-buffered through wave-private LDS with CDNA5 async
// global->LDS copies when the toolchain exposes them.
template <bool BIAS, bool RELU, bool HALF_OUT>
__global__ void __launch_bounds__(256) wmma_gemm_nt(
    const _Float16* __restrict__ A, const _Float16* __restrict__ B,
    const float* __restrict__ bias, void* __restrict__ C, int M, int N, int K) {
#if HAVE_ASYNC_LDS
  __shared__ __align__(16) _Float16 astage[8][2][16 * 32];
#endif
  const int wv = threadIdx.x >> 5;
  const int lane = threadIdx.x & 31;
  int wave = blockIdx.x * (blockDim.x >> 5) + wv;
  int tilesN = N >> 6;            // 64-wide output tiles
  int tm = wave / tilesN;
  int tn = wave - tm * tilesN;
  if ((tm << 4) >= M) return;
  const _Float16* arow = A + (size_t)(tm << 4) * K;
  const _Float16* brow = B + (size_t)(tn << 6) * K;
  v8f zero = {};
  v8f acc[4];
#pragma unroll
  for (int t = 0; t < 4; ++t) acc[t] = zero;

#if HAVE_ASYNC_LDS
  // Async copy of one 16x32 A chunk (16 rows x 64B) into LDS:
  // 2 x b128 per lane-group mapping: row = 8*half + (lane>>2), 16B chunk = lane&3.
  const int srow  = lane >> 2;
  const int sboff = (lane & 3) * 16;
  auto stageA = [&](int k0c, int buf) {
    const char* g0 = (const char*)(arow + k0c) + (size_t)srow * (K * 2) + sboff;
    const char* g1 = (const char*)(arow + k0c) + (size_t)(srow + 8) * (K * 2) + sboff;
    char* l0 = (char*)&astage[wv][buf][0] + srow * 64 + sboff;
    char* l1 = (char*)&astage[wv][buf][0] + (srow + 8) * 64 + sboff;
    __builtin_amdgcn_global_load_async_to_lds_b128((gptr4_t)(void*)g0,
                                                   (lptr4_t)(void*)l0, 0, 0);
    __builtin_amdgcn_global_load_async_to_lds_b128((gptr4_t)(void*)g1,
                                                   (lptr4_t)(void*)l1, 0, 0);
  };
  stageA(0, 0);
#endif

  for (int k0 = 0; k0 < K; k0 += 32) {
    __builtin_prefetch(brow + k0 + 32, 0, 1);
#if HAVE_ASYNC_LDS
    const int buf = (k0 >> 5) & 1;
    if (k0 + 32 < K) {
      stageA(k0 + 32, buf ^ 1);
      __builtin_amdgcn_s_wait_asynccnt(2);  // current buffer landed
    } else {
      __builtin_amdgcn_s_wait_asynccnt(0);
    }
    asm volatile("" ::: "memory");
    v16h af = ld_frag(&astage[wv][buf][0], 0, 15, 32, 0);
#else
    v16h af = ld_frag(arow, 0, 0x7fffffff, K, k0);
#endif
#pragma unroll
    for (int t = 0; t < 4; ++t) {
      v16h bf = ld_frag(brow + (size_t)(t << 4) * K, 0, 0x7fffffff, K, k0);
      acc[t] = wmma16(af, bf, acc[t]);
    }
  }

  int mbase = (tm << 4) + ((lane & 16) ? 8 : 0);
#pragma unroll
  for (int t = 0; t < 4; ++t) {
    int n = (tn << 6) + (t << 4) + (lane & 15);
    float bv = BIAS ? bias[n] : 0.0f;
#pragma unroll
    for (int r = 0; r < 8; ++r) {
      float val = acc[t][r] + bv;
      if (RELU) val = fmaxf(val, 0.0f);
      size_t idx = (size_t)(mbase + r) * N + n;
      if (HALF_OUT) ((_Float16*)C)[idx] = (_Float16)val;
      else          ((float*)C)[idx]    = val;
    }
  }
}

// ---------------------------------------------------- flash attention (XL)
// One wave = one 16-row query tile for one (batch, head). Streams keys in
// 32-wide steps with online softmax. Relative-position term uses the
// identity rel_shift: BD[i,j] = ((q_i+v) . rk[j + QLEN-1-i]) -> a 48-wide
// band per 32-key step, computed with WMMA and gathered through LDS.
__global__ void __launch_bounds__(256) attn_flash(
    const _Float16* __restrict__ q_h, const _Float16* __restrict__ k_h,
    const _Float16* __restrict__ vT_h, const _Float16* __restrict__ rk_h,
    const float* __restrict__ u, const float* __restrict__ v,
    _Float16* __restrict__ attno) {
  __shared__ __align__(16) float    band[8][16 * 48];
  __shared__ __align__(16) _Float16 ptile[8][16 * 32];
  const int ws   = threadIdx.x >> 5;
  const int lane = threadIdx.x & 31;
  const int jj   = lane & 15;
  const int hi8  = (lane & 16) ? 8 : 0;
  const int qt   = blockIdx.x * 8 + ws;
  const int bh   = blockIdx.y;
  const int b    = bh >> 3;
  const int h    = bh & 7;
  const int I0   = qt << 4;

  const _Float16* qp = q_h + (size_t)bh * QLEN * D_HEAD;
  const _Float16* kp = k_h + (size_t)bh * KLEN * D_HEAD;
  const _Float16* vp = vT_h + (size_t)bh * D_HEAD * KLEN;
  const _Float16* rp = rk_h + (size_t)h * KLEN * D_HEAD;

  // Q fragments with +u / +v folded in (two K-halves of d=64).
  v16h aU[2], aV[2];
  {
    const int row = I0 + jj;
#pragma unroll
    for (int f = 0; f < 2; ++f) {
      const _Float16* p = qp + (size_t)row * D_HEAD + f * 32 + hi8;
      v8h lo = *(const v8h*)(p);
      v8h hi = *(const v8h*)(p + 16);
      v16h uu, vv;
#pragma unroll
      for (int e = 0; e < 8; ++e) {
        int d0 = f * 32 + hi8 + e;
        int d1 = d0 + 16;
        float q0 = (float)lo[e], q1 = (float)hi[e];
        uu[e]     = (_Float16)(q0 + u[h * D_HEAD + d0]);
        uu[e + 8] = (_Float16)(q1 + u[h * D_HEAD + d1]);
        vv[e]     = (_Float16)(q0 + v[h * D_HEAD + d0]);
        vv[e + 8] = (_Float16)(q1 + v[h * D_HEAD + d1]);
      }
      aU[f] = uu; aV[f] = vv;
    }
  }

  v8f zero = {};
  v8f oacc[4];
#pragma unroll
  for (int t = 0; t < 4; ++t) oacc[t] = zero;
  float mrow[8], lrow[8];
#pragma unroll
  for (int r = 0; r < 8; ++r) { mrow[r] = -3.0e38f; lrow[r] = 0.0f; }

  int jEnd = I0 + 16 + MLEN;     // one past last unmasked key for this tile
  jEnd = (jEnd + 31) & ~31;
  if (jEnd > KLEN) jEnd = KLEN;

  for (int J0 = 0; J0 < jEnd; J0 += 32) {
    // ---- content scores AC for 32 keys (two 16-wide key tiles)
    v8f sc[2];
#pragma unroll
    for (int t = 0; t < 2; ++t) {
      v16h b0 = ld_frag(kp, J0 + 16 * t, KLEN - 1, D_HEAD, 0);
      v16h b1 = ld_frag(kp, J0 + 16 * t, KLEN - 1, D_HEAD, 32);
      v8f d = zero;
      d = wmma16(aU[0], b0, d);
      d = wmma16(aU[1], b1, d);
      sc[t] = d;
    }
    // ---- relative-position band: (q+v) @ rk[base .. base+47]^T
    const int base = J0 + QLEN - 16 - I0;  // >= 0 for all live tiles
#pragma unroll
    for (int bt = 0; bt < 3; ++bt) {
      v16h b0 = ld_frag(rp, base + 16 * bt, KLEN - 1, D_HEAD, 0);
      v16h b1 = ld_frag(rp, base + 16 * bt, KLEN - 1, D_HEAD, 32);
      v8f d = zero;
      d = wmma16(aV[0], b0, d);
      d = wmma16(aV[1], b1, d);
#pragma unroll
      for (int r = 0; r < 8; ++r)
        band[ws][(r + hi8) * 48 + 16 * bt + jj] = d[r];
    }
    asm volatile("s_wait_dscnt 0x0" ::: "memory");
    // ---- combine + mask + online softmax; stage P into LDS as f16
#pragma unroll
    for (int r = 0; r < 8; ++r) {
      const int ii = r + hi8;
      const int i = I0 + ii;
      float s0 = (sc[0][r] + band[ws][ii * 48 + (jj + 15 - ii)]) * 0.125f;
      float s1 = (sc[1][r] + band[ws][ii * 48 + 16 + (jj + 15 - ii)]) * 0.125f;
      if (J0 + jj > i + MLEN)      s0 = -3.0e38f;
      if (J0 + 16 + jj > i + MLEN) s1 = -3.0e38f;
      float mx = fmaxf(s0, s1);
      mx = fmaxf(mx, __shfl_xor(mx, 1));
      mx = fmaxf(mx, __shfl_xor(mx, 2));
      mx = fmaxf(mx, __shfl_xor(mx, 4));
      mx = fmaxf(mx, __shfl_xor(mx, 8));
      float mnew = fmaxf(mrow[r], mx);
      float corr = __expf(mrow[r] - mnew);
      mrow[r] = mnew;
      float p0 = __expf(s0 - mnew);
      float p1 = __expf(s1 - mnew);
      lrow[r] = lrow[r] * corr + p0 + p1;
#pragma unroll
      for (int t = 0; t < 4; ++t) oacc[t][r] *= corr;
      ptile[ws][ii * 32 + jj]      = (_Float16)p0;
      ptile[ws][ii * 32 + 16 + jj] = (_Float16)p1;
    }
    asm volatile("s_wait_dscnt 0x0" ::: "memory");
    // ---- P(16x32) @ V(32x64): relayout P C-tile -> A-fragment via LDS
    v16h pf;
    {
      const _Float16* pp = &ptile[ws][jj * 32 + hi8];
      v8h lo = *(const v8h*)(pp);
      v8h hi = *(const v8h*)(pp + 16);
      pf = mk16(lo, hi);
    }
#pragma unroll
    for (int t = 0; t < 4; ++t) {
      v16h vb = ld_frag(vp + (size_t)(t * 16) * KLEN, 0, 15, KLEN, J0);
      oacc[t] = wmma16(pf, vb, oacc[t]);
    }
  }

  // ---- normalize and store [qlen, B, n*d]
  float inv[8];
#pragma unroll
  for (int r = 0; r < 8; ++r) {
    float l = lrow[r];
    l += __shfl_xor(l, 1);
    l += __shfl_xor(l, 2);
    l += __shfl_xor(l, 4);
    l += __shfl_xor(l, 8);
    inv[r] = 1.0f / l;
  }
#pragma unroll
  for (int t = 0; t < 4; ++t)
#pragma unroll
    for (int r = 0; r < 8; ++r) {
      int i = I0 + r + hi8;
      size_t idx = ((size_t)i * BATCH + b) * NDH + h * D_HEAD + t * 16 + jj;
      attno[idx] = (_Float16)(oacc[t][r] * inv[r]);
    }
}

// -------------------------------------------------------------- add + LN
__global__ void __launch_bounds__(256) add_ln(
    const float* __restrict__ a, const float* __restrict__ c,
    const float* __restrict__ g, const float* __restrict__ be,
    float* __restrict__ out_f, _Float16* __restrict__ out_h) {
  const int row = blockIdx.x;
  const int tid = threadIdx.x;
  __shared__ float red[16];
  size_t base = (size_t)row * D_MODEL;
  float x0 = a[base + tid] + c[base + tid];
  float x1 = a[base + tid + 256] + c[base + tid + 256];
  float s = x0 + x1, sq = x0 * x0 + x1 * x1;
#pragma unroll
  for (int m = 1; m < 32; m <<= 1) { s += __shfl_xor(s, m); sq += __shfl_xor(sq, m); }
  if ((tid & 31) == 0) { red[tid >> 5] = s; red[8 + (tid >> 5)] = sq; }
  __syncthreads();
  if (tid < 32) {
    float ts = (tid < 8) ? red[tid] : 0.0f;
    float tq = (tid < 8) ? red[8 + tid] : 0.0f;
#pragma unroll
    for (int m = 1; m < 8; m <<= 1) { ts += __shfl_xor(ts, m); tq += __shfl_xor(tq, m); }
    if (tid == 0) { red[0] = ts; red[8] = tq; }
  }
  __syncthreads();
  float mean = red[0] * (1.0f / D_MODEL);
  float var  = red[8] * (1.0f / D_MODEL) - mean * mean;
  float rstd = rsqrtf(var + 1e-5f);
  float y0 = (x0 - mean) * rstd * g[tid] + be[tid];
  float y1 = (x1 - mean) * rstd * g[tid + 256] + be[tid + 256];
  out_f[base + tid] = y0;
  out_f[base + tid + 256] = y1;
  if (out_h) {
    out_h[base + tid] = (_Float16)y0;
    out_h[base + tid + 256] = (_Float16)y1;
  }
}

// ------------------------------------------------------------------ driver
extern "C" void kernel_launch(void* const* d_in, const int* in_sizes, int n_in,
                              void* d_out, int out_size, void* d_ws, size_t ws_size,
                              hipStream_t stream) {
  const float* x    = (const float*)d_in[0];
  const float* r    = (const float*)d_in[1];
  const float* u    = (const float*)d_in[2];
  const float* v    = (const float*)d_in[3];
  const float* mems = (const float*)d_in[4];
  const float* Wqkv = (const float*)d_in[5];
  const float* Wr   = (const float*)d_in[6];
  const float* Wo   = (const float*)d_in[7];
  const float* ln1g = (const float*)d_in[8];
  const float* ln1b = (const float*)d_in[9];
  const float* W1   = (const float*)d_in[10];
  const float* b1   = (const float*)d_in[11];
  const float* W2   = (const float*)d_in[12];
  const float* b2   = (const float*)d_in[13];
  const float* ln2g = (const float*)d_in[14];
  const float* ln2b = (const float*)d_in[15];
  (void)in_sizes; (void)n_in; (void)out_size; (void)ws_size;

  char* w = (char*)d_ws;
  size_t off = 0;
  auto alloc = [&](size_t bytes) {
    char* p = w + off;
    off = (off + bytes + 255) & ~(size_t)255;
    return p;
  };
  _Float16* cat_h  = (_Float16*)alloc((size_t)CROWS * D_MODEL * 2);
  _Float16* r_h    = (_Float16*)alloc((size_t)KLEN * D_MODEL * 2);
  _Float16* wqkv_h = (_Float16*)alloc((size_t)3 * NDH * D_MODEL * 2);
  _Float16* wr_h   = (_Float16*)alloc((size_t)NDH * D_MODEL * 2);
  _Float16* wo_h   = (_Float16*)alloc((size_t)D_MODEL * NDH * 2);
  _Float16* w1t_h  = (_Float16*)alloc((size_t)D_INNER * D_MODEL * 2);
  _Float16* w2t_h  = (_Float16*)alloc((size_t)D_MODEL * D_INNER * 2);
  _Float16* qkv_h  = (_Float16*)alloc((size_t)CROWS * 3 * NDH * 2);
  _Float16* rkf_h  = (_Float16*)alloc((size_t)KLEN * NDH * 2);
  _Float16* q_h    = (_Float16*)alloc((size_t)BATCH * N_HEADS * QLEN * D_HEAD * 2);
  _Float16* k_h    = (_Float16*)alloc((size_t)BATCH * N_HEADS * KLEN * D_HEAD * 2);
  _Float16* vT_h   = (_Float16*)alloc((size_t)BATCH * N_HEADS * D_HEAD * KLEN * 2);
  _Float16* rk_h   = (_Float16*)alloc((size_t)N_HEADS * KLEN * D_HEAD * 2);
  _Float16* ao_h   = (_Float16*)alloc((size_t)MROWS * NDH * 2);
  float*    o_f    = (float*)alloc((size_t)MROWS * D_MODEL * 4);
  float*    h_f    = (float*)alloc((size_t)MROWS * D_MODEL * 4);
  _Float16* h_h    = (_Float16*)alloc((size_t)MROWS * D_MODEL * 2);
  _Float16* ff1_h  = (_Float16*)alloc((size_t)MROWS * D_INNER * 2);
  float*    ff2_f  = (float*)alloc((size_t)MROWS * D_MODEL * 4);

  const int TB = 256;
  auto nb = [](long n) { return (int)((n + 255) / 256); };

  k_pack_cat<<<nb((long)CROWS * D_MODEL), TB, 0, stream>>>(x, mems, cat_h);
  k_cast<<<nb((long)KLEN * D_MODEL), TB, 0, stream>>>(r, r_h, KLEN * D_MODEL);
  k_cast<<<nb((long)3 * NDH * D_MODEL), TB, 0, stream>>>(Wqkv, wqkv_h, 3 * NDH * D_MODEL);
  k_cast<<<nb((long)NDH * D_MODEL), TB, 0, stream>>>(Wr, wr_h, NDH * D_MODEL);
  k_cast<<<nb((long)D_MODEL * NDH), TB, 0, stream>>>(Wo, wo_h, D_MODEL * NDH);
  k_transpose_cast<<<nb((long)D_MODEL * D_INNER), TB, 0, stream>>>(W1, w1t_h, D_MODEL, D_INNER);
  k_transpose_cast<<<nb((long)D_INNER * D_MODEL), TB, 0, stream>>>(W2, w2t_h, D_INNER, D_MODEL);

  // qkv = cat @ Wqkv^T  [8192 x 1536]
  wmma_gemm_nt<false, false, true>
      <<<(CROWS / 16) * (3 * NDH / 64) / 8, TB, 0, stream>>>(
          cat_h, wqkv_h, nullptr, qkv_h, CROWS, 3 * NDH, D_MODEL);
  // rk = r @ Wr^T  [4096 x 512]
  wmma_gemm_nt<false, false, true>
      <<<(KLEN / 16) * (NDH / 64) / 8, TB, 0, stream>>>(
          r_h, wr_h, nullptr, rkf_h, KLEN, NDH, D_MODEL);

  k_split_heads<<<nb((long)KLEN * BATCH * NDH), TB, 0, stream>>>(qkv_h, q_h, k_h, vT_h);
  k_split_rk<<<nb((long)KLEN * NDH), TB, 0, stream>>>(rkf_h, rk_h);

  dim3 ag(QLEN / 16 / 8, BATCH * N_HEADS);
  attn_flash<<<ag, TB, 0, stream>>>(q_h, k_h, vT_h, rk_h, u, v, ao_h);

  // out @ Wo^T  [4096 x 512] (f32)
  wmma_gemm_nt<false, false, false>
      <<<(MROWS / 16) * (D_MODEL / 64) / 8, TB, 0, stream>>>(
          ao_h, wo_h, nullptr, o_f, MROWS, D_MODEL, NDH);
  add_ln<<<MROWS, TB, 0, stream>>>(x, o_f, ln1g, ln1b, h_f, h_h);

  // FF1: relu(h @ W1 + b1)  [4096 x 2048]
  wmma_gemm_nt<true, true, true>
      <<<(MROWS / 16) * (D_INNER / 64) / 8, TB, 0, stream>>>(
          h_h, w1t_h, b1, ff1_h, MROWS, D_INNER, D_MODEL);
  // FF2: ff1 @ W2 + b2  [4096 x 512] (f32)
  wmma_gemm_nt<true, false, false>
      <<<(MROWS / 16) * (D_MODEL / 64) / 8, TB, 0, stream>>>(
          ff1_h, w2t_h, b2, ff2_f, MROWS, D_MODEL, D_INNER);

  add_ln<<<MROWS, TB, 0, stream>>>(h_f, ff2_f, ln2g, ln2b, (float*)d_out, nullptr);
}
